// MplugOwlVisionAttention_7232724926823
// MI455X (gfx1250) — matrix operand order repaired and checked by reading, not compiled, not run on hardware
//
#include <hip/hip_runtime.h>

typedef __attribute__((ext_vector_type(16))) _Float16 v16h;
typedef __attribute__((ext_vector_type(8)))  _Float16 v8h;
typedef __attribute__((ext_vector_type(8)))  float    v8f;

union V16H { v16h v; v8h h[2]; };

#define S_LEN   1025
#define BATCH   16
#define NHEAD   16
#define HDIM    64
#define HID     1024
#define N_QKV   3072
#define M_REAL  (BATCH * S_LEN)   /* 16400 */
#define M_PAD   16512             /* 129 * 128 */
#define SQ_PAD  1088              /* 17 * 64: all waves in a 4-wave block convergent */
#define SK_PAD  1056              /* 33 * 32  */

// ---------------------------------------------------------------------------
// Fragment loaders (gfx1250 wave32 WMMA; A and B both use contiguous k order)
// A-frag (16x32 f16, MxK): lane(g=l>>4, r=l&15) holds row M=r,
//   halves at K = g*16..+15 (32 contiguous bytes)
// B-frag (32x16 f16, KxN) from row-major [N,K]: lane holds col N=r,
//   halves at K = g*16..+15 (32 contiguous bytes)
// C/D element: (m = v + 8*g, n = r)
// ---------------------------------------------------------------------------
__device__ __forceinline__ v16h load_a_frag(const _Float16* p) {
  V16H u; u.h[0] = *(const v8h*)p; u.h[1] = *(const v8h*)(p + 8);
  return u.v;
}
__device__ __forceinline__ v16h load_b_frag(const _Float16* p) {
  V16H u; u.h[0] = *(const v8h*)p; u.h[1] = *(const v8h*)(p + 8);
  return u.v;
}

#define WMMA_F16(A, B, C) \
  __builtin_amdgcn_wmma_f32_16x16x32_f16(false, (A), false, (B), (short)0, (C), false, false)

// CDNA5 async global->LDS copy, 16B per lane, tracked by ASYNCcnt.
__device__ __forceinline__ void async_b128(_Float16* lds_dst, const _Float16* gsrc) {
  asm volatile("global_load_async_to_lds_b128 %0, %1, off"
               :: "v"((unsigned)(__SIZE_TYPE__)lds_dst), "v"(gsrc)
               : "memory");
}

// ---------------------------------------------------------------------------
// f32 -> f16 convert with zero tail padding
// ---------------------------------------------------------------------------
__global__ void cvt_f32_f16_kernel(const float* __restrict__ src,
                                   _Float16* __restrict__ dst,
                                   int n_src, int n_dst) {
  int i = blockIdx.x * 256 + threadIdx.x;
  if (i < n_dst) dst[i] = (i < n_src) ? (_Float16)src[i] : (_Float16)0.0f;
}

// ---------------------------------------------------------------------------
// QKV GEMM: C[M_PAD, 3072] = A[M_PAD,1024] x W[3072,1024]^T + bias
// 4 waves/block (2x2), wave tile 64x64: 16 WMMA per 16 b128 loads per K-step.
// Epilogue scatters to per-head Q (pre-scaled) / K / V^T f16 buffers.
// ---------------------------------------------------------------------------
__global__ __launch_bounds__(128) void qkv_gemm_kernel(
    const _Float16* __restrict__ A, const _Float16* __restrict__ W,
    const float* __restrict__ bias,
    _Float16* __restrict__ qb, _Float16* __restrict__ kb,
    _Float16* __restrict__ vtb)
{
  const int lane = threadIdx.x & 31;
  const int wid  = threadIdx.x >> 5;
  const int g = lane >> 4, r = lane & 15;
  const int bm = blockIdx.y * 128 + (wid >> 1) * 64;
  const int bn = blockIdx.x * 128 + (wid & 1) * 64;

  v8f acc[4][4];
  const v8f vz = {};
  #pragma unroll
  for (int mt = 0; mt < 4; ++mt)
    #pragma unroll
    for (int nt = 0; nt < 4; ++nt) acc[mt][nt] = vz;

  const _Float16* aptr[4];
  const _Float16* bptr[4];
  #pragma unroll
  for (int t = 0; t < 4; ++t) {
    aptr[t] = A + (size_t)(bm + t * 16 + r) * HID + g * 16;
    bptr[t] = W + (size_t)(bn + t * 16 + r) * HID + g * 16;
  }

  for (int k0 = 0; k0 < HID; k0 += 32) {
    v16h a[4];
    #pragma unroll
    for (int mt = 0; mt < 4; ++mt) a[mt] = load_a_frag(aptr[mt] + k0);
    #pragma unroll
    for (int nt = 0; nt < 4; ++nt) {
      v16h bf = load_b_frag(bptr[nt] + k0);
      #pragma unroll
      for (int mt = 0; mt < 4; ++mt)
        acc[mt][nt] = WMMA_F16(a[mt], bf, acc[mt][nt]);
    }
  }

  // Epilogue: o = head*192 + which*64 + d
  #pragma unroll
  for (int mt = 0; mt < 4; ++mt)
    #pragma unroll
    for (int nt = 0; nt < 4; ++nt)
      #pragma unroll
      for (int v = 0; v < 8; ++v) {
        int mrow = bm + mt * 16 + g * 8 + v;
        int ncol = bn + nt * 16 + r;
        if (mrow < M_REAL) {
          float val = acc[mt][nt][v] + bias[ncol];
          int bb   = mrow / S_LEN, s = mrow - bb * S_LEN;
          int head = ncol / 192,  rem = ncol - head * 192;
          int which = rem >> 6,   d = rem & 63;
          int bh = bb * NHEAD + head;
          if (which == 0)
            qb[((size_t)bh * SQ_PAD + s) * HDIM + d] = (_Float16)(val * 0.125f);
          else if (which == 1)
            kb[((size_t)bh * SK_PAD + s) * HDIM + d] = (_Float16)val;
          else
            vtb[((size_t)bh * HDIM + d) * SK_PAD + s] = (_Float16)val;
        }
      }
}

// ---------------------------------------------------------------------------
// Flash attention: 4 waves/block share double-buffered K/V^T chunks staged in
// LDS with global_load_async_to_lds_b128 (ASYNCcnt pipelined). Each wave owns
// one 16-row q tile; online softmax; P goes through a per-wave LDS tile to
// convert C-layout -> A-layout for the PV WMMA.
// ---------------------------------------------------------------------------
__global__ __launch_bounds__(128) void flash_attn_kernel(
    const _Float16* __restrict__ qb, const _Float16* __restrict__ kb,
    const _Float16* __restrict__ vtb, _Float16* __restrict__ ctx)
{
  __shared__ __align__(16) _Float16 Kl[2][32 * HDIM];  // [keys 32][d 64]
  __shared__ __align__(16) _Float16 Vl[2][HDIM * 32];  // [d 64][keys 32]
  __shared__ __align__(16) _Float16 Pl[4][16 * 32];    // per-wave P tile

  const int tid  = threadIdx.x;
  const int lane = tid & 31;
  const int wid  = tid >> 5;
  const int g = lane >> 4, r = lane & 15;
  const int h = blockIdx.y, b = blockIdx.z;
  const int bh = b * NHEAD + h;
  const int qt = blockIdx.x * 4 + wid;

  const _Float16* Q  = qb  + ((size_t)bh * SQ_PAD + qt * 16) * HDIM;
  const _Float16* K  = kb  + (size_t)bh * SK_PAD * HDIM;
  const _Float16* VT = vtb + (size_t)bh * HDIM * SK_PAD;

  // Q tile fragments (16 x 64 = 2 A-frags), softmax scale pre-folded into Q
  v16h aq0 = load_a_frag(Q + r * HDIM + 0  + g * 16);
  v16h aq1 = load_a_frag(Q + r * HDIM + 32 + g * 16);

  v8f ctxacc[4];
  const v8f vz = {};
  #pragma unroll
  for (int dt = 0; dt < 4; ++dt) ctxacc[dt] = vz;

  float m8[8], l8[8];
  #pragma unroll
  for (int v = 0; v < 8; ++v) { m8[v] = -1e30f; l8[v] = 0.0f; }

  // stage one 32-key chunk: K 32x64 (4KB) + V^T 64x32 (4KB): 4 async/thread
  auto stage = [&](int c, int buf) {
    const int kbase = c * 32;
    #pragma unroll
    for (int i = 0; i < 2; ++i) {
      int idx = tid + i * 128;
      int krow = idx >> 3, kcol = (idx & 7) * 8;
      async_b128(&Kl[buf][krow * HDIM + kcol],
                 K + (size_t)(kbase + krow) * HDIM + kcol);
      int vrow = idx >> 2, vcol = (idx & 3) * 8;
      async_b128(&Vl[buf][vrow * 32 + vcol],
                 VT + (size_t)vrow * SK_PAD + kbase + vcol);
    }
  };

  stage(0, 0);

  for (int c = 0; c < SK_PAD / 32; ++c) {
    const int kbase = c * 32;
    if (c + 1 < SK_PAD / 32) {
      stage(c + 1, (c + 1) & 1);
      asm volatile("s_wait_asynccnt 0x4" ::: "memory"); // chunk c landed, c+1 in flight
    } else {
      asm volatile("s_wait_asynccnt 0x0" ::: "memory");
    }
    __syncthreads();

    const _Float16* Kb = Kl[c & 1];
    const _Float16* Vb = Vl[c & 1];

    // scores: 16 q rows x 32 keys, contraction over d=64 (2 K-steps)
    v8f s0 = vz, s1 = vz;
    s0 = WMMA_F16(aq0, load_b_frag(Kb + (size_t)r * HDIM + 0  + g * 16), s0);
    s0 = WMMA_F16(aq1, load_b_frag(Kb + (size_t)r * HDIM + 32 + g * 16), s0);
    s1 = WMMA_F16(aq0, load_b_frag(Kb + (size_t)(16 + r) * HDIM + 0  + g * 16), s1);
    s1 = WMMA_F16(aq1, load_b_frag(Kb + (size_t)(16 + r) * HDIM + 32 + g * 16), s1);

    // mask padded keys (column index n == r per C-layout)
    if (kbase + r >= S_LEN) {
      #pragma unroll
      for (int v = 0; v < 8; ++v) s0[v] = -1e30f;
    }
    if (kbase + 16 + r >= S_LEN) {
      #pragma unroll
      for (int v = 0; v < 8; ++v) s1[v] = -1e30f;
    }

    // online softmax; row m = v + 8g lives across a 16-lane half
    float nm[8], alpha[8];
    #pragma unroll
    for (int v = 0; v < 8; ++v) {
      float mx = fmaxf(s0[v], s1[v]);
      #pragma unroll
      for (int off = 8; off; off >>= 1) mx = fmaxf(mx, __shfl_xor(mx, off, 32));
      nm[v] = fmaxf(m8[v], mx);
      alpha[v] = __expf(m8[v] - nm[v]);
      m8[v] = nm[v];
    }
    #pragma unroll
    for (int v = 0; v < 8; ++v) {
      float p0 = __expf(s0[v] - nm[v]);
      float p1 = __expf(s1[v] - nm[v]);
      Pl[wid][(g * 8 + v) * 32 + r]      = (_Float16)p0;
      Pl[wid][(g * 8 + v) * 32 + 16 + r] = (_Float16)p1;
      float t = p0 + p1;
      #pragma unroll
      for (int off = 8; off; off >>= 1) t += __shfl_xor(t, off, 32);
      l8[v] = l8[v] * alpha[v] + t;
    }
    #pragma unroll
    for (int dt = 0; dt < 4; ++dt)
      #pragma unroll
      for (int v = 0; v < 8; ++v) ctxacc[dt][v] *= alpha[v];

    // P (16x32) x V (32x64); per-wave P tile, DS ops in-order within wave
    v16h pa = load_a_frag(&Pl[wid][r * 32 + g * 16]);
    #pragma unroll
    for (int dt = 0; dt < 4; ++dt) {
      v16h bv = load_b_frag(Vb + (size_t)(dt * 16 + r) * 32 + g * 16);
      ctxacc[dt] = WMMA_F16(pa, bv, ctxacc[dt]);
    }

    __syncthreads();  // K/V LDS WAR before next chunk's async writes
  }

  // write ctx in [B, S, H] layout as f16 for the dense GEMM
  #pragma unroll
  for (int dt = 0; dt < 4; ++dt)
    #pragma unroll
    for (int v = 0; v < 8; ++v) {
      int srow = qt * 16 + g * 8 + v;
      if (srow < S_LEN) {
        float val = ctxacc[dt][v] / l8[v];
        ctx[((size_t)(b * S_LEN + srow)) * HID + h * HDIM + dt * 16 + r] = (_Float16)val;
      }
    }
}

// ---------------------------------------------------------------------------
// Dense GEMM: out[M,1024] = ctx[M_PAD,1024] x Wd[1024,1024]^T + bias (f32 out)
// ---------------------------------------------------------------------------
__global__ __launch_bounds__(128) void dense_gemm_kernel(
    const _Float16* __restrict__ A, const _Float16* __restrict__ W,
    const float* __restrict__ bias, float* __restrict__ out)
{
  const int lane = threadIdx.x & 31;
  const int wid  = threadIdx.x >> 5;
  const int g = lane >> 4, r = lane & 15;
  const int bm = blockIdx.y * 128 + (wid >> 1) * 64;
  const int bn = blockIdx.x * 128 + (wid & 1) * 64;

  v8f acc[4][4];
  const v8f vz = {};
  #pragma unroll
  for (int mt = 0; mt < 4; ++mt)
    #pragma unroll
    for (int nt = 0; nt < 4; ++nt) acc[mt][nt] = vz;

  const _Float16* aptr[4];
  const _Float16* bptr[4];
  #pragma unroll
  for (int t = 0; t < 4; ++t) {
    aptr[t] = A + (size_t)(bm + t * 16 + r) * HID + g * 16;
    bptr[t] = W + (size_t)(bn + t * 16 + r) * HID + g * 16;
  }

  for (int k0 = 0; k0 < HID; k0 += 32) {
    v16h a[4];
    #pragma unroll
    for (int mt = 0; mt < 4; ++mt) a[mt] = load_a_frag(aptr[mt] + k0);
    #pragma unroll
    for (int nt = 0; nt < 4; ++nt) {
      v16h bf = load_b_frag(bptr[nt] + k0);
      #pragma unroll
      for (int mt = 0; mt < 4; ++mt)
        acc[mt][nt] = WMMA_F16(a[mt], bf, acc[mt][nt]);
    }
  }

  #pragma unroll
  for (int mt = 0; mt < 4; ++mt)
    #pragma unroll
    for (int nt = 0; nt < 4; ++nt)
      #pragma unroll
      for (int v = 0; v < 8; ++v) {
        int mrow = bm + mt * 16 + g * 8 + v;
        int ncol = bn + nt * 16 + r;
        if (mrow < M_REAL)
          out[(size_t)mrow * HID + ncol] = acc[mt][nt][v] + bias[ncol];
      }
}

// ---------------------------------------------------------------------------
extern "C" void kernel_launch(void* const* d_in, const int* in_sizes, int n_in,
                              void* d_out, int out_size, void* d_ws, size_t ws_size,
                              hipStream_t stream)
{
  const float* hs   = (const float*)d_in[0];   // [16, 1025, 1024]
  const float* wqkv = (const float*)d_in[1];   // [3072, 1024]
  const float* bqkv = (const float*)d_in[2];   // [3072]
  const float* wdns = (const float*)d_in[3];   // [1024, 1024]
  const float* bdns = (const float*)d_in[4];   // [1024]
  float* out = (float*)d_out;                  // [16, 1025, 1024]

  char* ws = (char*)d_ws;
  size_t off = 0;
  auto take = [&](size_t bytes) -> char* {
    char* p = ws + off;
    off += (bytes + 255) & ~(size_t)255;
    return p;
  };

  const size_t szA  = (size_t)M_PAD * HID * 2;
  const size_t szWq = (size_t)N_QKV * HID * 2;
  const size_t szWd = (size_t)HID * HID * 2;
  const size_t szQ  = (size_t)BATCH * NHEAD * SQ_PAD * HDIM * 2;
  const size_t szK  = (size_t)BATCH * NHEAD * SK_PAD * HDIM * 2;
  const size_t szVT = (size_t)BATCH * NHEAD * HDIM * SK_PAD * 2;
  const size_t szC  = (size_t)M_PAD * HID * 2;

  _Float16* A16  = (_Float16*)take(szA);
  _Float16* Wq16 = (_Float16*)take(szWq);
  _Float16* Wd16 = (_Float16*)take(szWd);
  _Float16* Q16  = (_Float16*)take(szQ);
  _Float16* K16  = (_Float16*)take(szK);
  _Float16* VT16 = (_Float16*)take(szVT);
  _Float16* C16  = (_Float16*)take(szC);

  // zero pads (graph-capture-safe async memsets)
  hipMemsetAsync(Q16,  0, szQ,  stream);
  hipMemsetAsync(K16,  0, szK,  stream);
  hipMemsetAsync(VT16, 0, szVT, stream);
  hipMemsetAsync(C16,  0, szC,  stream);

  // f32 -> f16 conversions (A padded with zeros to M_PAD rows)
  {
    int n_src = M_REAL * HID, n_dst = M_PAD * HID;
    cvt_f32_f16_kernel<<<(n_dst + 255) / 256, 256, 0, stream>>>(hs, A16, n_src, n_dst);
  }
  {
    int n = N_QKV * HID;
    cvt_f32_f16_kernel<<<(n + 255) / 256, 256, 0, stream>>>(wqkv, Wq16, n, n);
  }
  {
    int n = HID * HID;
    cvt_f32_f16_kernel<<<(n + 255) / 256, 256, 0, stream>>>(wdns, Wd16, n, n);
  }

  // fused QKV projection + per-head scatter (Q pre-scaled, V transposed)
  qkv_gemm_kernel<<<dim3(N_QKV / 128, M_PAD / 128), 128, 0, stream>>>(
      A16, Wq16, bqkv, Q16, K16, VT16);

  // flash attention (async double-buffered K/V^T staging in LDS)
  flash_attn_kernel<<<dim3(SQ_PAD / 64, NHEAD, BATCH), 128, 0, stream>>>(
      Q16, K16, VT16, C16);

  // output projection
  dense_gemm_kernel<<<dim3(HID / 128, M_PAD / 128), 128, 0, stream>>>(
      C16, Wd16, bdns, out);
}
